// DMHSA_86861418594903
// MI455X (gfx1250) — matrix operand order repaired and checked by reading, not compiled, not run on hardware
//
#include <hip/hip_runtime.h>
#include <hip/hip_bf16.h>

// ---------------------------------------------------------------------------
// DMHSA for MI455X (gfx1250), wave32, f16 WMMA with f32 accumulate.
// Pipeline: convert -> fused QKV projection GEMM (WMMA) ->
//           fused attention kernel: QK^T (WMMA) -> masked softmax in LDS ->
//           PV (WMMA from LDS), attn written to HBM exactly once.
// ---------------------------------------------------------------------------

typedef __attribute__((ext_vector_type(16))) _Float16 v16h;
typedef __attribute__((ext_vector_type(8)))  _Float16 v8h;
typedef __attribute__((ext_vector_type(4)))  _Float16 v4h;
typedef __attribute__((ext_vector_type(8)))  float    v8f;
typedef __attribute__((ext_vector_type(4)))  float    v4f;

#define BATCH 4
#define SEQ   2048
#define CDIM  1024
#define HEADS 16
#define HDIM  64
#define BH    (BATCH * HEADS)
#define MTOT  (BATCH * SEQ)      // 8192
#define SCALE 0.03125f           // C^-0.5 = 1/32

// fused attention tiling
#define RROWS      16            // rows of scores resident in LDS per block
#define LDS_STRIDE 2052          // 2048 + 4 floats pad (8-row offset = 32 banks)

static __device__ __forceinline__ v8f wmma_f16(v16h a, v16h b, v8f c) {
    return __builtin_amdgcn_wmma_f32_16x16x32_f16(
        /*neg_a=*/false, a, /*neg_b=*/false, b,
        /*c_mod=*/(short)0, c, /*reuse_a=*/false, /*reuse_b=*/false);
}

// A-matrix 16x32 f16 fragment (row-major source, ld in halves).
// Lane L: row = row0 + (L%16); halves at k0+(L>>4)*8 (v0..3) and +16 (v4..7).
static __device__ __forceinline__ v16h load_a_frag(const _Float16* __restrict__ base,
                                                   int ld, int row0, int k0, int lane) {
    int r = row0 + (lane & 15);
    int kb = k0 + ((lane >> 4) << 3);
    const _Float16* p = base + (size_t)r * ld + kb;
    v8h lo = *(const v8h*)(p);
    v8h hi = *(const v8h*)(p + 16);
    v16h f;
#pragma unroll
    for (int i = 0; i < 8; ++i) { f[i] = lo[i]; f[8 + i] = hi[i]; }
    return f;
}

// A fragment from f32 LDS rows (attn stripe); convert on the fly.
static __device__ __forceinline__ v16h lds_a_frag_f32(const float* base, int ld,
                                                      int k0, int lane) {
    int r = lane & 15;
    int kb = k0 + ((lane >> 4) << 3);
    const float* p = base + r * ld + kb;
    v4f x0 = *(const v4f*)(p);
    v4f x1 = *(const v4f*)(p + 4);
    v4f y0 = *(const v4f*)(p + 16);
    v4f y1 = *(const v4f*)(p + 20);
    v16h f;
#pragma unroll
    for (int i = 0; i < 4; ++i) {
        f[i]      = (_Float16)x0[i];
        f[4 + i]  = (_Float16)x1[i];
        f[8 + i]  = (_Float16)y0[i];
        f[12 + i] = (_Float16)y1[i];
    }
    return f;
}

// B-matrix 32x16 f16 fragment; source stored as B^T row-major: [col, k], k contiguous.
// Lane L: col = col0 + (L%16); halves at k0+(L>>4)*8 (v0..3) and +16 (v4..7).
static __device__ __forceinline__ v16h load_b_frag(const _Float16* __restrict__ base,
                                                   int ld, int col0, int k0, int lane) {
    int c = col0 + (lane & 15);
    int kb = k0 + ((lane >> 4) << 3);
    const _Float16* p = base + (size_t)c * ld + kb;
    v8h lo = *(const v8h*)(p);
    v8h hi = *(const v8h*)(p + 16);
    v16h f;
#pragma unroll
    for (int i = 0; i < 8; ++i) { f[i] = lo[i]; f[8 + i] = hi[i]; }
    return f;
}

static __device__ __forceinline__ float wave_sum(float x) {
#pragma unroll
    for (int off = 16; off > 0; off >>= 1) x += __shfl_xor(x, off, 32);
    return x;
}
static __device__ __forceinline__ float wave_max(float x) {
#pragma unroll
    for (int off = 16; off > 0; off >>= 1) x = fmaxf(x, __shfl_xor(x, off, 32));
    return x;
}

// ---------------------------------------------------------------------------
// Pass 0: f32 -> f16 conversion (4 elements per thread)
// ---------------------------------------------------------------------------
__global__ void cvt_f32_f16_kernel(const float* __restrict__ src,
                                   _Float16* __restrict__ dst, int n4) {
    int i = blockIdx.x * blockDim.x + threadIdx.x;
    if (i < n4) {
        v4f v = *(const v4f*)(src + (size_t)i * 4);
        v4h h;
#pragma unroll
        for (int j = 0; j < 4; ++j) h[j] = (_Float16)v[j];
        *(v4h*)(dst + (size_t)i * 4) = h;
    }
}

// ---------------------------------------------------------------------------
// Pass 1: fused projection GEMM  y = x @ W^T  (+ bias for V)
//   grid = (1024, 3), block = 256 (8 waves). Each wave: 32x32 output tile.
//   which = 0:Q -> [b,h,n,d], 1:K -> [b,h,n,d], 2:V(+bv) -> transposed [b,h,d,n]
// ---------------------------------------------------------------------------
__global__ __launch_bounds__(256)
void proj_kernel(const _Float16* __restrict__ xh, const _Float16* __restrict__ wh,
                 const float* __restrict__ bv,
                 _Float16* __restrict__ qh, _Float16* __restrict__ kh,
                 _Float16* __restrict__ vth) {
    const int which = blockIdx.y;
    const _Float16* W = wh + (size_t)which * CDIM * CDIM;
    const int wl   = blockIdx.x * 8 + (threadIdx.x >> 5);  // 0..8191 wave tiles
    const int lane = threadIdx.x & 31;
    const int m0 = (wl >> 5) * 32;   // 256 row tiles
    const int n0 = (wl & 31) * 32;   // 32 col tiles

    v8f acc00 = {}, acc01 = {}, acc10 = {}, acc11 = {};
    for (int k0 = 0; k0 < CDIM; k0 += 32) {
        v16h a0 = load_a_frag(xh, CDIM, m0,      k0, lane);
        v16h a1 = load_a_frag(xh, CDIM, m0 + 16, k0, lane);
        v16h b0 = load_b_frag(W,  CDIM, n0,      k0, lane);
        v16h b1 = load_b_frag(W,  CDIM, n0 + 16, k0, lane);
        acc00 = wmma_f16(a0, b0, acc00);
        acc01 = wmma_f16(a0, b1, acc01);
        acc10 = wmma_f16(a1, b0, acc10);
        acc11 = wmma_f16(a1, b1, acc11);
    }

    const int lr = (lane >> 4) << 3;   // row sub-offset 0 or 8
    const int lc = lane & 15;
#pragma unroll
    for (int i = 0; i < 2; ++i) {
#pragma unroll
        for (int j = 0; j < 2; ++j) {
            const v8f& acc = (i == 0) ? (j == 0 ? acc00 : acc01)
                                      : (j == 0 ? acc10 : acc11);
            const int n = n0 + j * 16 + lc;
            const int hh = n >> 6, d = n & 63;
#pragma unroll
            for (int v = 0; v < 8; ++v) {
                const int m = m0 + i * 16 + lr + v;
                const int bidx = m >> 11, nrow = m & (SEQ - 1);
                float val = acc[v];
                if (which == 2) {
                    val += bv[n];
                    vth[(((size_t)bidx * HEADS + hh) * HDIM + d) * SEQ + nrow] = (_Float16)val;
                } else {
                    size_t idx = (((size_t)bidx * HEADS + hh) * SEQ + nrow) * HDIM + d;
                    if (which == 0) qh[idx] = (_Float16)val;
                    else            kh[idx] = (_Float16)val;
                }
            }
        }
    }
}

// ---------------------------------------------------------------------------
// Pass 2: fused attention. One block (256 thr, 8 waves) per (bh, 16-row stripe).
//   LDS: scores stripe [16][2052] f32 + PV partials [2][16*64] + rowinv[16].
//   Phase 1: S = scale * Q K^T          (WMMA, -> LDS)
//   Phase 2: dynamic-mask softmax rows  (attn -> HBM once; e kept in LDS)
//   Phase 3: out = (1/sum) * e @ V      (WMMA from LDS)
// ---------------------------------------------------------------------------
__global__ __launch_bounds__(256)
void attn_fused_kernel(const _Float16* __restrict__ qh, const _Float16* __restrict__ kh,
                       const _Float16* __restrict__ vth,
                       float* __restrict__ attn, float* __restrict__ out) {
    extern __shared__ float smem[];
    float* S      = smem;                            // [RROWS][LDS_STRIDE]
    float* OB     = smem + RROWS * LDS_STRIDE;       // [2][RROWS*HDIM]
    float* ROWINV = OB + 2 * RROWS * HDIM;           // [RROWS]

    const int bh = blockIdx.y;
    const int b  = bh >> 4, h = bh & 15;
    const int m0 = blockIdx.x * RROWS;
    const int w    = threadIdx.x >> 5;
    const int lane = threadIdx.x & 31;
    const int lr = (lane >> 4) << 3;
    const int lc = lane & 15;

    const _Float16* q  = qh  + (size_t)bh * SEQ * HDIM;
    const _Float16* k  = kh  + (size_t)bh * SEQ * HDIM;
    const _Float16* Vt = vth + (size_t)bh * HDIM * SEQ;

    // ---------------- Phase 1: score stripe into LDS ----------------
    {
        v16h aq0 = load_a_frag(q, HDIM, m0, 0,  lane);   // rows m0..m0+15, k 0..31
        v16h aq1 = load_a_frag(q, HDIM, m0, 32, lane);   // rows m0..m0+15, k 32..63
        for (int mt = w; mt < SEQ / 32; mt += 8) {
            const int n0 = mt * 32;
            v16h b0 = load_b_frag(k, HDIM, n0,      0,  lane);
            v16h b1 = load_b_frag(k, HDIM, n0 + 16, 0,  lane);
            v16h b2 = load_b_frag(k, HDIM, n0,      32, lane);
            v16h b3 = load_b_frag(k, HDIM, n0 + 16, 32, lane);
            v8f acc0 = {}, acc1 = {};
            acc0 = wmma_f16(aq0, b0, acc0);
            acc0 = wmma_f16(aq1, b2, acc0);
            acc1 = wmma_f16(aq0, b1, acc1);
            acc1 = wmma_f16(aq1, b3, acc1);
#pragma unroll
            for (int v = 0; v < 8; ++v) {
                S[(lr + v) * LDS_STRIDE + n0 + lc]      = acc0[v] * SCALE;
                S[(lr + v) * LDS_STRIDE + n0 + 16 + lc] = acc1[v] * SCALE;
            }
        }
    }
    __syncthreads();

    // ---------------- Phase 2: masked softmax (wave w: rows 2w, 2w+1) --------
#pragma unroll
    for (int rr = 0; rr < 2; ++rr) {
        const int row = w * 2 + rr;
        float* Sr = S + row * LDS_STRIDE;

        float sum = 0.f, mx = -3.4e38f;
        for (int cc = 0; cc < SEQ; cc += 128) {
            v4f sv = *(const v4f*)(Sr + cc + lane * 4);
#pragma unroll
            for (int j = 0; j < 4; ++j) { sum += sv[j]; mx = fmaxf(mx, sv[j]); }
        }
        sum = wave_sum(sum);
        mx  = wave_max(mx);
        const float mean = sum * (1.0f / SEQ);

        float esum = 0.f;
        for (int cc = 0; cc < SEQ; cc += 128) {
            v4f sv = *(const v4f*)(Sr + cc + lane * 4);
            v4f e;
#pragma unroll
            for (int j = 0; j < 4; ++j) {
                e[j] = (sv[j] >= mean) ? __expf(sv[j] - mx) : 0.f;
                esum += e[j];
            }
            *(v4f*)(Sr + cc + lane * 4) = e;   // keep unnormalized e for PV
        }
        esum = wave_sum(esum);
        const float inv = 1.0f / esum;
        if (lane == 0) ROWINV[row] = inv;

        // single HBM write of the normalized attn row
        float* arow = attn + ((size_t)bh * SEQ + m0 + row) * SEQ;
        for (int cc = 0; cc < SEQ; cc += 128) {
            v4f e = *(const v4f*)(Sr + cc + lane * 4);
            v4f o;
#pragma unroll
            for (int j = 0; j < 4; ++j) o[j] = e[j] * inv;
            *(v4f*)(arow + cc + lane * 4) = o;
        }
    }
    __syncthreads();

    // ---------------- Phase 3: PV from LDS ----------------
    {
        const int ct  = w & 3;          // d-tile: 4 x 16 cols
        const int khf = w >> 1 >> 1;    // k half: 0 or 1
        const int d0  = ct * 16;
        v8f acc = {};
        const int kbeg = khf * (SEQ / 2);
        for (int k0 = kbeg; k0 < kbeg + SEQ / 2; k0 += 32) {
            v16h af = lds_a_frag_f32(S, LDS_STRIDE, k0, lane);
            v16h bf = load_b_frag(Vt, SEQ, d0, k0, lane);
            acc = wmma_f16(af, bf, acc);
        }
        float* P = OB + khf * (RROWS * HDIM);
#pragma unroll
        for (int v = 0; v < 8; ++v)
            P[(lr + v) * HDIM + d0 + lc] = acc[v];
    }
    __syncthreads();

    // combine k-halves, scale by 1/rowsum, write out[b, m0+row, h*64+d]
    {
        const int idx = threadIdx.x * 4;          // RROWS*HDIM = 1024 = 256*4
        const int row = idx >> 6, d = idx & 63;
        v4f o;
#pragma unroll
        for (int j = 0; j < 4; ++j)
            o[j] = (OB[idx + j] + OB[RROWS * HDIM + idx + j]) * ROWINV[row];
        *(v4f*)(out + ((size_t)b * SEQ + m0 + row) * CDIM + h * HDIM + d) = o;
    }
}

// ---------------------------------------------------------------------------
extern "C" void kernel_launch(void* const* d_in, const int* in_sizes, int n_in,
                              void* d_out, int out_size, void* d_ws, size_t ws_size,
                              hipStream_t stream) {
    const float* x  = (const float*)d_in[0];  // [4, 2048, 1024]
    const float* Wq = (const float*)d_in[1];  // [1024, 1024]
    const float* Wk = (const float*)d_in[2];
    const float* Wv = (const float*)d_in[3];
    const float* bv = (const float*)d_in[4];  // [1024]

    float* out_p  = (float*)d_out;                     // [4, 2048, 1024]
    float* attn_p = out_p + (size_t)MTOT * CDIM;       // [4, 16, 2048, 2048]

    _Float16* ws  = (_Float16*)d_ws;
    _Float16* xh  = ws;                                      // 8388608 halves
    _Float16* wh  = xh + (size_t)MTOT * CDIM;                // 3 * 1048576
    _Float16* qh  = wh + (size_t)3 * CDIM * CDIM;            // 8388608
    _Float16* kh  = qh + (size_t)BH * SEQ * HDIM;            // 8388608
    _Float16* vth = kh + (size_t)BH * SEQ * HDIM;            // 8388608
    (void)in_sizes; (void)n_in; (void)out_size; (void)ws_size;

    // Pass 0: convert inputs to f16
    {
        int n4 = (MTOT * CDIM) / 4;
        cvt_f32_f16_kernel<<<(n4 + 255) / 256, 256, 0, stream>>>(x, xh, n4);
        int w4 = (CDIM * CDIM) / 4;
        cvt_f32_f16_kernel<<<(w4 + 255) / 256, 256, 0, stream>>>(Wq, wh, w4);
        cvt_f32_f16_kernel<<<(w4 + 255) / 256, 256, 0, stream>>>(Wk, wh + (size_t)CDIM * CDIM, w4);
        cvt_f32_f16_kernel<<<(w4 + 255) / 256, 256, 0, stream>>>(Wv, wh + (size_t)2 * CDIM * CDIM, w4);
    }
    // Pass 1: Q/K/V projections (WMMA)
    proj_kernel<<<dim3(1024, 3), 256, 0, stream>>>(xh, wh, bv, qh, kh, vth);
    // Pass 2: fused QK^T + masked softmax + PV (WMMA + LDS-resident stripe)
    const size_t smem_bytes =
        (size_t)(RROWS * LDS_STRIDE + 2 * RROWS * HDIM + RROWS) * sizeof(float);
    attn_fused_kernel<<<dim3(SEQ / RROWS, BH), 256, smem_bytes, stream>>>(
        qh, kh, vth, attn_p, out_p);
}
